// CodecNormalizer_40664750359336
// MI455X (gfx1250) — matrix operand order repaired and checked by reading, not compiled, not run on hardware
//
#include <hip/hip_runtime.h>
#include <hip/hip_bf16.h>
#include <stdint.h>

// ---------- types for CDNA5 WMMA ----------
typedef __attribute__((ext_vector_type(16))) __bf16 v16bf;
typedef __attribute__((ext_vector_type(8)))  float  v8f;
typedef __attribute__((ext_vector_type(4)))  float  v4f;   // native vector: OK for nontemporal builtins

// float -> bf16 (RNE, NaN-safe), stored as ushort
__device__ __forceinline__ unsigned short f2bf(float f) {
    unsigned int u = __float_as_uint(f);
    if ((u & 0x7fffffffu) > 0x7f800000u) return (unsigned short)((u >> 16) | 0x40); // quiet NaN
    u += 0x7fffu + ((u >> 16) & 1u);
    return (unsigned short)(u >> 16);
}

// ---------- 1) temporal mean pool -> bf16 activations ----------
// x viewed as [M, T, 32] v4f ; out: [M, 128] bf16.
// grid = M blocks, block = 128 threads (4 waves, each owns a T/4 slice).
// Non-temporal loads: 536MB one-shot stream must not evict L2-resident weights.
__global__ void pool_mean_bf16(const v4f* __restrict__ x, unsigned short* __restrict__ out, int T) {
    const int m    = blockIdx.x;
    const int lane = threadIdx.x & 31;
    const int w    = threadIdx.x >> 5;      // wave id 0..3
    const int tPer = T >> 2;                // rows per wave
    const v4f* p = x + (size_t)m * T * 32 + (size_t)w * tPer * 32 + lane;
    v4f s = {0.f, 0.f, 0.f, 0.f};
    for (int t = 0; t < tPer; ++t) {
        v4f v = __builtin_nontemporal_load(&p[(size_t)t * 32]);
        s += v;
    }
    __shared__ v4f sd[4][32];
    sd[w][lane] = s;
    __syncthreads();
    if (w == 0) {
        v4f tot = sd[0][lane] + sd[1][lane] + sd[2][lane] + sd[3][lane];
        const float inv = 1.0f / (float)T;
        unsigned short o0 = f2bf(tot.x * inv);
        unsigned short o1 = f2bf(tot.y * inv);
        unsigned short o2 = f2bf(tot.z * inv);
        unsigned short o3 = f2bf(tot.w * inv);
        uint2 pk;
        pk.x = (unsigned)o0 | ((unsigned)o1 << 16);
        pk.y = (unsigned)o2 | ((unsigned)o3 << 16);
        *(uint2*)&out[(size_t)m * 128 + lane * 4] = pk;
    }
}

// ---------- 2) weight transpose + convert: W[K,N] f32 -> Wt[N,K] bf16 ----------
__global__ void transpose_to_bf16(const float* __restrict__ W, unsigned short* __restrict__ Wt,
                                  int K, int N) {
    int i = blockIdx.x * blockDim.x + threadIdx.x;
    if (i >= K * N) return;
    int n = i / K, k = i - n * K;
    Wt[(size_t)n * K + k] = f2bf(W[(size_t)k * N + n]);
}

// ---------- 3) GEMM: out[M,N] = act(A[M,K](bf16) x Wt[N,K]^T(bf16) + bias) ----------
// One wave32 per 16x32 output tile: A fragment reused across two WMMA.
// grid = (N/32, M/16), block = 32.
__global__ void gemm_bf16_wmma(const unsigned short* __restrict__ A,
                               const unsigned short* __restrict__ Bt,
                               const float* __restrict__ bias,
                               void* __restrict__ out,
                               int N, int K, int relu, int store_bf16) {
    const int lane = threadIdx.x & 31;
    const int half = lane >> 4;       // K-half selector
    const int l15  = lane & 15;       // row (A) / col (B,C)
    const int m0 = blockIdx.y * 16;
    const int n0 = blockIdx.x * 32;

    v8f c0 = {}, c1 = {};
    const unsigned short* arow  = A  + (size_t)(m0 + l15) * K;
    const unsigned short* brow0 = Bt + (size_t)(n0 + l15) * K + 16 * half;        // 32B-aligned
    const unsigned short* brow1 = Bt + (size_t)(n0 + 16 + l15) * K + 16 * half;

    for (int kb = 0; kb < K; kb += 32) {
        // A fragment: VGPR j holds K = 2j + 16*half, pair-contiguous (matches B's contiguous K order)
        union { v16bf v; unsigned int u[8]; } af;
#pragma unroll
        for (int j = 0; j < 8; ++j) {
            int koff = 2 * j + 16 * half;
            af.u[j] = *(const unsigned int*)(arow + kb + koff);   // vectorizes to 2x b128
        }
        // B fragments: lane's 16 contiguous K values (K = 16*half + e) from transposed weights
        v16bf b0 = *(const v16bf*)(brow0 + kb);                   // one 32B load
        v16bf b1 = *(const v16bf*)(brow1 + kb);
        c0 = __builtin_amdgcn_wmma_f32_16x16x32_bf16(false, af.v, false, b0,
                                                     (short)0, c0, false, false);
        c1 = __builtin_amdgcn_wmma_f32_16x16x32_bf16(false, af.v, false, b1,
                                                     (short)0, c1, false, false);
    }

    const int nA = n0 + l15;
    const int nB = n0 + 16 + l15;
    const float bvA = bias[nA];
    const float bvB = bias[nB];
#pragma unroll
    for (int r = 0; r < 8; ++r) {
        int m = m0 + r + 8 * half;            // ISA C/D layout: VGPR r -> M=r (+8 for upper lanes)
        float vA = c0[r] + bvA;
        float vB = c1[r] + bvB;
        if (relu) { vA = fmaxf(vA, 0.f); vB = fmaxf(vB, 0.f); }
        if (store_bf16) {
            ((unsigned short*)out)[(size_t)m * N + nA] = f2bf(vA);
            ((unsigned short*)out)[(size_t)m * N + nB] = f2bf(vB);
        } else {
            ((float*)out)[(size_t)m * N + nA] = vA;
            ((float*)out)[(size_t)m * N + nB] = vB;
        }
    }
}

// ---------- 4) row-wise L2 normalize, E: [M,256] f32 in place ----------
__global__ void l2norm_rows(float* __restrict__ E) {
    __shared__ float red[256];
    const int r = blockIdx.x, t = threadIdx.x;
    float v = E[(size_t)r * 256 + t];
    red[t] = v * v;
    __syncthreads();
    for (int s = 128; s > 0; s >>= 1) {
        if (t < s) red[t] += red[t + s];
        __syncthreads();
    }
    E[(size_t)r * 256 + t] = v * __frsqrt_rn(red[0]);
}

// ---------- 5) argmin of pairwise L1 distance via packed u64 atomicMin ----------
__global__ void init_keys(unsigned long long* keys) {
    if (threadIdx.x < 16) keys[threadIdx.x] = ~0ull;
}

__global__ void l1_argmin(const float* __restrict__ dEmb,   // [16,256] normalized
                          const float* __restrict__ nEmb,   // [4096,256] normalized
                          unsigned long long* __restrict__ keys) {
    __shared__ float sd[16 * 256];
    for (int i = threadIdx.x; i < 16 * 256; i += blockDim.x) sd[i] = dEmb[i];
    __syncthreads();
    const int n = blockIdx.x * blockDim.x + threadIdx.x;    // 0..4095
    float acc[16];
#pragma unroll
    for (int b = 0; b < 16; ++b) acc[b] = 0.f;
    const v4f* row = (const v4f*)(nEmb + (size_t)n * 256);
    for (int c4 = 0; c4 < 64; ++c4) {
        v4f x = row[c4];
        const int c = c4 * 4;
#pragma unroll
        for (int b = 0; b < 16; ++b) {
            const float* s = &sd[b * 256 + c];
            acc[b] += fabsf(x.x - s[0]) + fabsf(x.y - s[1])
                    + fabsf(x.z - s[2]) + fabsf(x.w - s[3]);
        }
    }
#pragma unroll
    for (int b = 0; b < 16; ++b) {
        unsigned long long key = ((unsigned long long)__float_as_uint(acc[b]) << 32)
                               | (unsigned long long)(unsigned)n;
        atomicMin(&keys[b], key);
    }
}

// ---------- 6) gather winning codec rows: out[b] = normal[idx_b]  (256*128 f32) ----------
__global__ void gather_rows(const unsigned long long* __restrict__ keys,
                            const float* __restrict__ normal, float* __restrict__ out) {
    const int b = blockIdx.y;
    const unsigned idx = (unsigned)(keys[b] & 0xffffffffull);
    const v4f* src = (const v4f*)(normal + (size_t)idx * 32768);
    v4f*       dst = (v4f*)(out + (size_t)b * 32768);
    const int i = blockIdx.x * blockDim.x + threadIdx.x;    // 8192 v4f per row
    dst[i] = src[i];
}

// =====================================================================
extern "C" void kernel_launch(void* const* d_in, const int* in_sizes, int n_in,
                              void* d_out, int out_size, void* d_ws, size_t ws_size,
                              hipStream_t stream) {
    const float* dys = (const float*)d_in[0];   // [16,512,128]
    const float* nrm = (const float*)d_in[1];   // [4096,256,128]
    const float* W1  = (const float*)d_in[2];   // [128,512]
    const float* b1  = (const float*)d_in[3];
    const float* W2  = (const float*)d_in[4];   // [512,512]
    const float* b2  = (const float*)d_in[5];
    const float* W3  = (const float*)d_in[6];   // [512,256]
    const float* b3  = (const float*)d_in[7];
    float* out = (float*)d_out;

    const int B = 16, N = 4096, Td = 512, Tn = 256, D = 128, H = 512, E = 256;

    // workspace carve-up (256B aligned)
    uintptr_t p = ((uintptr_t)d_ws + 255) & ~(uintptr_t)255;
    auto take = [&](size_t bytes) { uintptr_t q = p; p = (p + bytes + 255) & ~(uintptr_t)255; return (void*)q; };
    unsigned short* Xn  = (unsigned short*)take((size_t)N * D * 2);  // pooled normal, bf16
    unsigned short* Xd  = (unsigned short*)take((size_t)B * D * 2);  // pooled dysarthric, bf16
    unsigned short* W1t = (unsigned short*)take((size_t)H * D * 2);  // [512,128]
    unsigned short* W2t = (unsigned short*)take((size_t)H * H * 2);  // [512,512]
    unsigned short* W3t = (unsigned short*)take((size_t)E * H * 2);  // [256,512]
    unsigned short* H1n = (unsigned short*)take((size_t)N * H * 2);
    unsigned short* H1d = (unsigned short*)take((size_t)B * H * 2);
    unsigned short* H2n = (unsigned short*)take((size_t)N * H * 2);
    unsigned short* H2d = (unsigned short*)take((size_t)B * H * 2);
    float*          En  = (float*)take((size_t)N * E * 4);
    float*          Ed  = (float*)take((size_t)B * E * 4);
    unsigned long long* keys = (unsigned long long*)take(16 * 8);

    // 1) mean pool (bandwidth-bound 536MB NT stream) + bf16 convert
    pool_mean_bf16<<<N, 128, 0, stream>>>((const v4f*)nrm, Xn, Tn);
    pool_mean_bf16<<<B, 128, 0, stream>>>((const v4f*)dys, Xd, Td);

    // 2) weight transpose/convert (once per launch; tiny, stays hot in L2)
    transpose_to_bf16<<<(D * H + 255) / 256, 256, 0, stream>>>(W1, W1t, D, H);
    transpose_to_bf16<<<(H * H + 255) / 256, 256, 0, stream>>>(W2, W2t, H, H);
    transpose_to_bf16<<<(H * E + 255) / 256, 256, 0, stream>>>(W3, W3t, H, E);

    // 3) MLP via v_wmma_f32_16x16x32_bf16 (one wave per 16x32 tile, A-frag reuse)
    // layer 1: [M,128] x [128,512] + b1, relu
    gemm_bf16_wmma<<<dim3(H / 32, N / 16), 32, 0, stream>>>(Xn, W1t, b1, H1n, H, D, 1, 1);
    gemm_bf16_wmma<<<dim3(H / 32, B / 16), 32, 0, stream>>>(Xd, W1t, b1, H1d, H, D, 1, 1);
    // layer 2: [M,512] x [512,512] + b2, relu
    gemm_bf16_wmma<<<dim3(H / 32, N / 16), 32, 0, stream>>>(H1n, W2t, b2, H2n, H, H, 1, 1);
    gemm_bf16_wmma<<<dim3(H / 32, B / 16), 32, 0, stream>>>(H1d, W2t, b2, H2d, H, H, 1, 1);
    // layer 3: [M,512] x [512,256] + b3, fp32 out
    gemm_bf16_wmma<<<dim3(E / 32, N / 16), 32, 0, stream>>>(H2n, W3t, b3, En, E, H, 0, 0);
    gemm_bf16_wmma<<<dim3(E / 32, B / 16), 32, 0, stream>>>(H2d, W3t, b3, Ed, E, H, 0, 0);

    // 4) L2 normalize embeddings
    l2norm_rows<<<N, 256, 0, stream>>>(En);
    l2norm_rows<<<B, 256, 0, stream>>>(Ed);

    // 5) L1 distance + argmin (packed u64 atomicMin, re-initialized every call)
    init_keys<<<1, 32, 0, stream>>>(keys);
    l1_argmin<<<N / 256, 256, 0, stream>>>(Ed, En, keys);

    // 6) gather winners into d_out [16,256,128]
    gather_rows<<<dim3(32, B), 256, 0, stream>>>(keys, nrm, out);

    (void)in_sizes; (void)n_in; (void)out_size; (void)ws_size;
}
